// Attention_55284819034164
// MI455X (gfx1250) — matrix operand order repaired and checked by reading, not compiled, not run on hardware
//
#include <hip/hip_runtime.h>

// Multi-head attention (flash style) + output projection for MI455X (gfx1250).
//   B=4, L=1024, D=1024, H=16, HEAD_DIM=64, scale = 1/sqrt(D) = 1/32.
//
// v2 changes vs v1 (driven by histogram: valu=413, vmem=176 vs wmma=12):
//  * K/V tiles staged in LDS as f16, cooperatively by the whole block, double
//    buffered: f32->f16 conversion done ONCE per tile instead of once per wave
//    (4x less cvt VALU), global traffic /4, V gather becomes ds_load_b128.
//  * V stored transposed in LDS (Vt[d][key]) with 72-half row stride so both
//    the transposed stores and the fragment loads are LDS-bank-conflict free.
//  * Software pipelining: next tile's global loads issue before compute, cvt +
//    ds_store after; one barrier per iteration; global_prefetch_b8 for j0+64.
//  * W_H pre-converted to f16 once; projection inner loop has zero cvt and
//    reuses each W fragment for two row tiles (8 WMMA : 12 b128 loads).

#define Bc  4
#define Lc  1024
#define Dc  1024
#define Hc  16
#define HDc 64

#define KSTR 72   // LDS row stride in halfs for Kh (32 rows x 64 used)
#define VSTR 72   // LDS row stride in halfs for Vt (64 rows x 32 used)

typedef __attribute__((ext_vector_type(16))) _Float16 v16h;
typedef __attribute__((ext_vector_type(8)))  _Float16 v8h;
typedef __attribute__((ext_vector_type(2)))  _Float16 v2h;
typedef __attribute__((ext_vector_type(8)))  float    v8f;

static __device__ __forceinline__ v8f wmma_f16(v16h a, v16h b, v8f c) {
  // D = A(16x32,f16) * B(32x16,f16) + C(16x16,f32)
  return __builtin_amdgcn_wmma_f32_16x16x32_f16(
      false, a, false, b, (short)0, c, false, false);
}

template <int BASE>
static __device__ __forceinline__ void put8(v16h& a, float4 p, float4 q, float s) {
  a[BASE + 0] = (_Float16)(p.x * s); a[BASE + 1] = (_Float16)(p.y * s);
  a[BASE + 2] = (_Float16)(p.z * s); a[BASE + 3] = (_Float16)(p.w * s);
  a[BASE + 4] = (_Float16)(q.x * s); a[BASE + 5] = (_Float16)(q.y * s);
  a[BASE + 6] = (_Float16)(q.z * s); a[BASE + 7] = (_Float16)(q.w * s);
}

// B-fragment chunk: 16 consecutive f32 -> 16 f16 (k = e + 16*hi pattern).
static __device__ __forceinline__ v16h load16f(const float* p, float s) {
  float4 a = *(const float4*)(p + 0);
  float4 b = *(const float4*)(p + 4);
  float4 c = *(const float4*)(p + 8);
  float4 d = *(const float4*)(p + 12);
  v16h v;
  put8<0>(v, a, b, s);
  put8<8>(v, c, d, s);
  return v;
}

// Two runs of 8 consecutive halfs -> one 16-half fragment register set.
static __device__ __forceinline__ v16h load8x2h(const _Float16* p0, const _Float16* p1) {
  v8h a = *(const v8h*)p0;
  v8h b = *(const v8h*)p1;
  return __builtin_shufflevector(a, b, 0, 1, 2, 3, 4, 5, 6, 7,
                                 8, 9, 10, 11, 12, 13, 14, 15);
}

static __device__ __forceinline__ v8h cvt8(float4 a, float4 b) {
  v8h h;
  h[0] = (_Float16)a.x; h[1] = (_Float16)a.y; h[2] = (_Float16)a.z; h[3] = (_Float16)a.w;
  h[4] = (_Float16)b.x; h[5] = (_Float16)b.y; h[6] = (_Float16)b.z; h[7] = (_Float16)b.w;
  return h;
}

// Packed transposed V store: Vt[d][2*kp .. 2*kp+1] = {a, b} as one b32.
static __device__ __forceinline__ void stv(_Float16* vbuf, int d, int kp, float a, float b) {
  v2h p;
  p[0] = (_Float16)a;
  p[1] = (_Float16)b;
  *(v2h*)(vbuf + d * VSTR + 2 * kp) = p;
}

// ---------------------------------------------------------------------------
// Kernel 1: flash attention. Block = 128 threads (4 waves), all on one (b,h);
// wave w owns q rows [q0, q0+16). K/V tiles (32 keys x 64 dims) are staged in
// LDS as f16 (V transposed), double buffered. Grid = (L/64, H, B).
// Writes softmax(QK^T/32)V to workspace X as f16 [B, L, D].
// ---------------------------------------------------------------------------
__global__ __launch_bounds__(128) void attn_fwd_kernel(
    const float* __restrict__ Q, const float* __restrict__ K,
    const float* __restrict__ V, _Float16* __restrict__ X) {
  __shared__ _Float16 Kh[2][32 * KSTR];  // [key][d]
  __shared__ _Float16 Vt[2][64 * VSTR];  // [d][key] (transposed)

  const int tid  = threadIdx.x;
  const int lane = tid & 31;
  const int wave = tid >> 5;
  const int l16  = lane & 15;
  const int hi   = lane >> 4;
  const int h    = blockIdx.y;
  const int b    = blockIdx.z;
  const int q0   = blockIdx.x * 64 + wave * 16;

  // Staging assignments (cover the full 32x64 tile with 128 threads).
  const int kr  = tid & 31;   // K: row (key), 16 floats at col ks*16
  const int ks  = tid >> 5;
  const int vkp = tid & 15;   // V: key pair (2*vkp, 2*vkp+1), 8 dims at vdg*8
  const int vdg = tid >> 4;

  const size_t bL = (size_t)b * Lc;
  const float* gK  = K + (bL + kr) * Dc + h * HDc + ks * 16;
  const float* gV0 = V + (bL + 2 * vkp + 0) * Dc + h * HDc + vdg * 8;
  const float* gV1 = V + (bL + 2 * vkp + 1) * Dc + h * HDc + vdg * 8;

  const float scale = 0.03125f;  // 1/sqrt(1024), folded into Q

  // Q^T as B-fragments (k = head dim, n = q row): two 32-deep chunks.
  const float* qrow = Q + (bL + q0 + l16) * Dc + h * HDc;
  const v16h bQ0 = load16f(qrow + 0 + 16 * hi, scale);
  const v16h bQ1 = load16f(qrow + 32 + 16 * hi, scale);

  // Prologue: stage tile j0=0 into buffer 0.
  {
    float4 k0 = *(const float4*)(gK + 0), k1 = *(const float4*)(gK + 4);
    float4 k2 = *(const float4*)(gK + 8), k3 = *(const float4*)(gK + 12);
    float4 a0 = *(const float4*)(gV0 + 0), a1 = *(const float4*)(gV0 + 4);
    float4 b0 = *(const float4*)(gV1 + 0), b1 = *(const float4*)(gV1 + 4);
    _Float16* kd = &Kh[0][kr * KSTR + ks * 16];
    *(v8h*)(kd + 0) = cvt8(k0, k1);
    *(v8h*)(kd + 8) = cvt8(k2, k3);
    _Float16* vd = &Vt[0][0];
    stv(vd, vdg * 8 + 0, vkp, a0.x, b0.x);
    stv(vd, vdg * 8 + 1, vkp, a0.y, b0.y);
    stv(vd, vdg * 8 + 2, vkp, a0.z, b0.z);
    stv(vd, vdg * 8 + 3, vkp, a0.w, b0.w);
    stv(vd, vdg * 8 + 4, vkp, a1.x, b1.x);
    stv(vd, vdg * 8 + 5, vkp, a1.y, b1.y);
    stv(vd, vdg * 8 + 6, vkp, a1.z, b1.z);
    stv(vd, vdg * 8 + 7, vkp, a1.w, b1.w);
  }

  v8f o0 = {}, o1 = {}, o2 = {}, o3 = {};
  float mrun = -__builtin_inff();
  float lrun = 0.0f;
  int s = 0;

  for (int j0 = 0; j0 < Lc; j0 += 32) {
    __syncthreads();  // stage s visible to all waves; buffer s^1 free to write

    // Issue next tile's global loads up front (consumed after compute).
    const bool more = (j0 + 32) < Lc;
    float4 k0, k1, k2, k3, a0, a1, vb0, vb1;
    if (more) {
      const size_t adv = (size_t)(j0 + 32) * Dc;
      const float* pk = gK + adv;
      k0 = *(const float4*)(pk + 0); k1 = *(const float4*)(pk + 4);
      k2 = *(const float4*)(pk + 8); k3 = *(const float4*)(pk + 12);
      const float* pa = gV0 + adv;
      const float* pb = gV1 + adv;
      a0 = *(const float4*)(pa + 0); a1 = *(const float4*)(pa + 4);
      vb0 = *(const float4*)(pb + 0); vb1 = *(const float4*)(pb + 4);
      if (j0 + 64 < Lc) {  // pull tile j0+64 toward L2 (global_prefetch_b8)
        const size_t adv2 = (size_t)(j0 + 64) * Dc;
        __builtin_prefetch(gK + adv2, 0, 1);
        __builtin_prefetch(gV0 + adv2, 0, 1);
      }
    }

    // ---- scores: S^T[m=key][n=q] for two 16-key subtiles, from LDS f16 ----
    v8f s0 = {}, s1 = {};
    {
      const _Float16* kp0 = &Kh[s][(0 + l16) * KSTR];
      v16h f0 = load8x2h(kp0 + 16 * hi, kp0 + 16 * hi + 8);
      v16h f1 = load8x2h(kp0 + 32 + 16 * hi, kp0 + 32 + 16 * hi + 8);
      s0 = wmma_f16(f0, bQ0, s0);
      s0 = wmma_f16(f1, bQ1, s0);
      const _Float16* kp1 = &Kh[s][(16 + l16) * KSTR];
      v16h g0 = load8x2h(kp1 + 16 * hi, kp1 + 16 * hi + 8);
      v16h g1 = load8x2h(kp1 + 32 + 16 * hi, kp1 + 32 + 16 * hi + 8);
      s1 = wmma_f16(g0, bQ0, s1);
      s1 = wmma_f16(g1, bQ1, s1);
    }

    // ---- online softmax (rows = q = l16 per lane) ----
    float tmax = -__builtin_inff();
#pragma unroll
    for (int r = 0; r < 8; ++r) {
      tmax = fmaxf(tmax, s0[r]);
      tmax = fmaxf(tmax, s1[r]);
    }
    tmax = fmaxf(tmax, __shfl_xor(tmax, 16, 32));
    const float nm = fmaxf(mrun, tmax);
    const float alpha = __expf(mrun - nm);
    mrun = nm;

    float rsum = 0.0f;
    v16h aP;  // P^T accumulator layout == A-fragment layout for P*V
#pragma unroll
    for (int r = 0; r < 8; ++r) {
      const float p0 = __expf(s0[r] - nm);
      const float p1 = __expf(s1[r] - nm);
      rsum += p0 + p1;
      aP[r]     = (_Float16)p0;
      aP[r + 8] = (_Float16)p1;
    }
    rsum += __shfl_xor(rsum, 16, 32);
    lrun = lrun * alpha + rsum;

    v8f aB;  // alpha broadcast into C-layout rows (m = r + 8*hi)
#pragma unroll
    for (int r = 0; r < 8; ++r) aB[r] = __shfl(alpha, r + 8 * hi, 32);
    o0 *= aB; o1 *= aB; o2 *= aB; o3 *= aB;

    // ---- O += P*V, V B-fragments straight from transposed LDS tile ----
    {
      const _Float16* v0 = &Vt[s][(0  + l16) * VSTR + 8 * hi];
      const _Float16* v1 = &Vt[s][(16 + l16) * VSTR + 8 * hi];
      const _Float16* v2 = &Vt[s][(32 + l16) * VSTR + 8 * hi];
      const _Float16* v3 = &Vt[s][(48 + l16) * VSTR + 8 * hi];
      o0 = wmma_f16(aP, load8x2h(v0, v0 + 16), o0);
      o1 = wmma_f16(aP, load8x2h(v1, v1 + 16), o1);
      o2 = wmma_f16(aP, load8x2h(v2, v2 + 16), o2);
      o3 = wmma_f16(aP, load8x2h(v3, v3 + 16), o3);
    }

    // ---- convert + store next tile into buffer s^1 ----
    if (more) {
      _Float16* kd = &Kh[s ^ 1][kr * KSTR + ks * 16];
      *(v8h*)(kd + 0) = cvt8(k0, k1);
      *(v8h*)(kd + 8) = cvt8(k2, k3);
      _Float16* vd = &Vt[s ^ 1][0];
      stv(vd, vdg * 8 + 0, vkp, a0.x, vb0.x);
      stv(vd, vdg * 8 + 1, vkp, a0.y, vb0.y);
      stv(vd, vdg * 8 + 2, vkp, a0.z, vb0.z);
      stv(vd, vdg * 8 + 3, vkp, a0.w, vb0.w);
      stv(vd, vdg * 8 + 4, vkp, a1.x, vb1.x);
      stv(vd, vdg * 8 + 5, vkp, a1.y, vb1.y);
      stv(vd, vdg * 8 + 6, vkp, a1.z, vb1.z);
      stv(vd, vdg * 8 + 7, vkp, a1.w, vb1.w);
    }
    s ^= 1;
  }

  // Finalize: divide rows by softmax denominator, store f16 activations.
  const float linv = 1.0f / lrun;
  v8f lB;
#pragma unroll
  for (int r = 0; r < 8; ++r) lB[r] = __shfl(linv, r + 8 * hi, 32);
  o0 *= lB; o1 *= lB; o2 *= lB; o3 *= lB;

#pragma unroll
  for (int r = 0; r < 8; ++r) {
    _Float16* xp = X + (bL + q0 + r + 8 * hi) * Dc + h * HDc + l16;
    xp[0]  = (_Float16)o0[r];
    xp[16] = (_Float16)o1[r];
    xp[32] = (_Float16)o2[r];
    xp[48] = (_Float16)o3[r];
  }
}

// ---------------------------------------------------------------------------
// Kernel 1b: one-time W_H f32 -> f16 conversion (amortizes all projection cvt).
// ---------------------------------------------------------------------------
__global__ __launch_bounds__(256) void cvt_w_kernel(
    const float* __restrict__ W, _Float16* __restrict__ Wh) {
  const int i = (blockIdx.x * 256 + threadIdx.x) * 8;
  float4 a = *(const float4*)(W + i);
  float4 b = *(const float4*)(W + i + 4);
  *(v8h*)(Wh + i) = cvt8(a, b);
}

// ---------------------------------------------------------------------------
// Kernel 2: Out = X @ W^T + bias, X and W in f16. One wave computes a 32x64
// tile (2 row tiles reuse each W fragment). Grid = (B*L/32, D/256), block 128.
// ---------------------------------------------------------------------------
__global__ __launch_bounds__(128) void out_proj_kernel(
    const _Float16* __restrict__ X, const _Float16* __restrict__ Wh,
    const float* __restrict__ bias, float* __restrict__ Out) {
  const int lane = threadIdx.x & 31;
  const int wave = threadIdx.x >> 5;
  const int l16  = lane & 15;
  const int hi   = lane >> 4;
  const int m0   = blockIdx.x * 32;               // two 16-row tiles
  const int n0   = (blockIdx.y * 4 + wave) * 64;  // 64 output cols per wave

  v8f c00, c01, c02, c03, c10, c11, c12, c13;
#pragma unroll
  for (int c = 0; c < 4; ++c) {
    const float bb = bias[n0 + c * 16 + l16];
    v8f t;
#pragma unroll
    for (int r = 0; r < 8; ++r) t[r] = bb;
    if (c == 0) { c00 = t; c10 = t; }
    if (c == 1) { c01 = t; c11 = t; }
    if (c == 2) { c02 = t; c12 = t; }
    if (c == 3) { c03 = t; c13 = t; }
  }

  const _Float16* x0 = X + ((size_t)(m0 + l16)) * Dc;
  const _Float16* x1 = X + ((size_t)(m0 + 16 + l16)) * Dc;
  const _Float16* w0 = Wh + ((size_t)(n0 + 0  + l16)) * Dc;
  const _Float16* w1 = Wh + ((size_t)(n0 + 16 + l16)) * Dc;
  const _Float16* w2 = Wh + ((size_t)(n0 + 32 + l16)) * Dc;
  const _Float16* w3 = Wh + ((size_t)(n0 + 48 + l16)) * Dc;

  for (int d0 = 0; d0 < Dc; d0 += 32) {
    const v16h aX0 = load8x2h(x0 + d0 + 16 * hi, x0 + d0 + 16 * hi + 8);
    const v16h aX1 = load8x2h(x1 + d0 + 16 * hi, x1 + d0 + 16 * hi + 8);
    const v16h bw0 = load8x2h(w0 + d0 + 16 * hi, w0 + d0 + 16 * hi + 8);
    const v16h bw1 = load8x2h(w1 + d0 + 16 * hi, w1 + d0 + 16 * hi + 8);
    const v16h bw2 = load8x2h(w2 + d0 + 16 * hi, w2 + d0 + 16 * hi + 8);
    const v16h bw3 = load8x2h(w3 + d0 + 16 * hi, w3 + d0 + 16 * hi + 8);
    c00 = wmma_f16(aX0, bw0, c00);
    c10 = wmma_f16(aX1, bw0, c10);
    c01 = wmma_f16(aX0, bw1, c01);
    c11 = wmma_f16(aX1, bw1, c11);
    c02 = wmma_f16(aX0, bw2, c02);
    c12 = wmma_f16(aX1, bw2, c12);
    c03 = wmma_f16(aX0, bw3, c03);
    c13 = wmma_f16(aX1, bw3, c13);
  }

#pragma unroll
  for (int r = 0; r < 8; ++r) {
    float* p0 = Out + ((size_t)(m0 + r + 8 * hi)) * Dc + n0 + l16;
    p0[0] = c00[r]; p0[16] = c01[r]; p0[32] = c02[r]; p0[48] = c03[r];
    float* p1 = Out + ((size_t)(m0 + 16 + r + 8 * hi)) * Dc + n0 + l16;
    p1[0] = c10[r]; p1[16] = c11[r]; p1[32] = c12[r]; p1[48] = c13[r];
  }
}

extern "C" void kernel_launch(void* const* d_in, const int* in_sizes, int n_in,
                              void* d_out, int out_size, void* d_ws, size_t ws_size,
                              hipStream_t stream) {
  (void)in_sizes; (void)n_in; (void)out_size; (void)ws_size;
  const float* Q  = (const float*)d_in[0];
  const float* K  = (const float*)d_in[1];
  const float* V  = (const float*)d_in[2];
  const float* WH = (const float*)d_in[3];
  const float* bH = (const float*)d_in[4];
  // d_in[5] is the (falsy) mask scalar -> no masking branch.
  float* out = (float*)d_out;

  _Float16* X  = (_Float16*)d_ws;                       // B*L*D f16 = 8 MB
  _Float16* Wh = (_Float16*)((char*)d_ws + (size_t)Bc * Lc * Dc * 2);  // 2 MB

  cvt_w_kernel<<<dim3((Dc * Dc) / (256 * 8)), 256, 0, stream>>>(WH, Wh);

  dim3 g1(Lc / 64, Hc, Bc);
  attn_fwd_kernel<<<g1, 128, 0, stream>>>(Q, K, V, X);

  dim3 g2((Bc * Lc) / 32, Dc / 256);
  out_proj_kernel<<<g2, 128, 0, stream>>>(X, Wh, bH, out);
}
